// _CRU_46875273068601
// MI455X (gfx1250) — matrix operand order repaired and checked by reading, not compiled, not run on hardware
//
#include <hip/hip_runtime.h>
#include <hip/hip_bf16.h>

// ---------------------------------------------------------------------------
// CRU (GRU-like, 3 parallel sub-states) for MI455X / gfx1250.
//   B=128, T=2048, IN=256, HID=256, OUT=256, gates packed 3*HID=768.
// Latency-bound serial recurrence -> persistent kernel + per-step grid
// barrier; all GEMMs on v_wmma_f32_16x16x32_bf16.
// 96 WGs x 128 threads: 1 wave per SIMD -> minimal per-step WMMA chain.
// All 48 time-invariant B-operand tiles live in VGPRs (~550 VGPRs/wave,
// legal at 1 wave/SIMD): the steady-state loop has NO LDS traffic at all.
// ---------------------------------------------------------------------------

#define B_SZ   128
#define T_SZ   2048
#define IN_D   256
#define HID_D  256
#define G3     768          // 3*HID, packed gate dim (r,z,n)
#define OUT_D  256
#define NWG    96           // 3 k * 16 col-slices (16 cols) * 2 row-halves (64 rows)
#define HELEMS (3 * B_SZ * HID_D)   // one h buffer [3,B,H]

typedef __attribute__((ext_vector_type(16))) __bf16 v16bf;
typedef __attribute__((ext_vector_type(8)))  float  v8f;

__device__ __forceinline__ v8f wmma_bf16(v16bf a, v16bf b, v8f c) {
  // (neg_a, A, neg_b, B, c_mod, C, reuse_a, reuse_b)
  return __builtin_amdgcn_wmma_f32_16x16x32_bf16(false, a, false, b,
                                                 (short)0, c, false, false);
}

__device__ __forceinline__ v8f zero8() {
  v8f z;
#pragma unroll
  for (int i = 0; i < 8; ++i) z[i] = 0.0f;
  return z;
}

// A-operand (16x32 bf16) gather: caller's pointer is the start of this lane's
// f32 row at the current K offset. ISA 16-bit A layout: lanes 0-15 hold
// K=0..7 (VGPR0-3) and K=16..23 (VGPR4-7); lanes 16-31 hold +8.
__device__ __forceinline__ v16bf load_a_row(const float* __restrict__ p, int half) {
  v16bf a;
#pragma unroll
  for (int v = 0; v < 8; ++v) {
    const int kk = ((v < 4) ? 2 * v : 16 + 2 * (v - 4)) + 8 * half;
    const float2 f = *(const float2*)(p + kk);
    a[2 * v]     = (__bf16)f.x;
    a[2 * v + 1] = (__bf16)f.y;
  }
  return a;
}

__device__ __forceinline__ float sigmoidf_(float x) {
  return 1.0f / (1.0f + __expf(-x));
}

// Sense-reversal grid barrier with monotonic phase (re-armed by init kernel
// each launch -> graph-replay safe).
__device__ __forceinline__ void grid_barrier(unsigned* cnt, unsigned* sense,
                                             unsigned phase) {
  __threadfence();          // release this wave's h stores to device scope
  __syncthreads();
  if (threadIdx.x == 0) {
    unsigned prev = __hip_atomic_fetch_add(cnt, 1u, __ATOMIC_ACQ_REL,
                                           __HIP_MEMORY_SCOPE_AGENT);
    if (prev == NWG - 1) {
      __hip_atomic_store(cnt, 0u, __ATOMIC_RELAXED, __HIP_MEMORY_SCOPE_AGENT);
      __hip_atomic_store(sense, phase, __ATOMIC_RELEASE,
                         __HIP_MEMORY_SCOPE_AGENT);
    } else {
      while (__hip_atomic_load(sense, __ATOMIC_ACQUIRE,
                               __HIP_MEMORY_SCOPE_AGENT) != phase) {
        __builtin_amdgcn_s_sleep(2);
      }
    }
  }
  __syncthreads();
  __threadfence();          // acquire: invalidate stale h lines
}

// ---------------------------------------------------------------------------
// init: copy h0 into h buffer 0, arm the barrier.
// ---------------------------------------------------------------------------
__global__ void cru_init(const float* __restrict__ h0, float* __restrict__ hb0,
                         unsigned* __restrict__ bar) {
  const int i = blockIdx.x * 256 + threadIdx.x;
  if (i < HELEMS) hb0[i] = h0[i];
  if (i < 4) bar[i] = 0u;
}

// ---------------------------------------------------------------------------
// Persistent recurrence. WG = (k, 16-col hidden slice, 64-row batch half).
// 4 waves (1 per SIMD) cover the 64 batch rows. Weight slices are staged
// through LDS once (bf16, exact WMMA B-operand lane order), then hoisted
// into VGPRs for the whole time loop.
// Bp[mat][gate][kstep][lane][e], K = kstep*32 + 8*(lane>>4) + e + 8*(e>=8), N = lane&15.
// ---------------------------------------------------------------------------
__global__ __launch_bounds__(128, 1)
void cru_persistent(const float* __restrict__ x,   // [B,T,IN]
                    const float* __restrict__ Wx,  // [3,IN,768]
                    const float* __restrict__ Wh,  // [3,HID,768]
                    const float* __restrict__ bx,  // [3,768]
                    const float* __restrict__ bh,  // [3,768]
                    float* __restrict__ hb0,       // [3,B,H] double buffer A
                    float* __restrict__ hb1,       // [3,B,H] double buffer B
                    unsigned* __restrict__ bar) {
  __shared__ alignas(32) __bf16 Bp[2][3][8][32][16];   // 48 KB staging

  const int tid     = threadIdx.x;
  const int k       = blockIdx.x >> 5;          // 0..2: sub-state matrix
  const int slice   = (blockIdx.x >> 1) & 15;   // hidden-column slice
  const int rowHalf = blockIdx.x & 1;           // batch rows [0,64) or [64,128)
  const int j0      = slice * 16;

  // -------- one-time: pack weight slices into LDS (bf16, operand order) ----
  const float* Wxk = Wx + (size_t)k * IN_D  * G3;
  const float* Whk = Wh + (size_t)k * HID_D * G3;
  for (int f = tid; f < 2 * 3 * 8 * 32 * 16; f += 128) {
    const int e    = f & 15;
    const int l    = (f >> 4) & 31;
    const int ks   = (f >> 9) & 7;
    const int g    = (f >> 12) % 3;
    const int mat  = f / 12288;
    const int K    = ks * 32 + (l >> 4) * 8 + e + ((e >= 8) ? 8 : 0);
    const int col  = g * HID_D + j0 + (l & 15);
    const float* src = (mat == 0) ? Wxk : Whk;
    Bp[mat][g][ks][l][e] = (__bf16)src[(size_t)K * G3 + col];
  }
  __syncthreads();

  const int lane = tid & 31;
  const int wave = tid >> 5;                    // 0..3, one per SIMD32
  const int half = lane >> 4;
  const int nlo  = lane & 15;
  const int brow = rowHalf * 64 + wave * 16 + nlo;  // A row this lane feeds
  const int j    = j0 + nlo;                        // output hidden column

  // -------- hoist ALL time-invariant B operands into VGPRs -----------------
  // 48 x v16bf = 384 VGPRs; fine at 1 wave/SIMD. Steady-state loop then has
  // zero LDS traffic and the WMMA chain never waits on dscnt.
  v16bf breg[2][3][8];
#pragma unroll
  for (int mat = 0; mat < 2; ++mat)
#pragma unroll
    for (int g = 0; g < 3; ++g)
#pragma unroll
      for (int ks = 0; ks < 8; ++ks)
        breg[mat][g][ks] = *(const v16bf*)&Bp[mat][g][ks][lane][0];

  // Per-lane gate biases (constant over time).
  const float rb  = bx[k * G3 + j]             + bh[k * G3 + j];
  const float zb  = bx[k * G3 + HID_D + j]     + bh[k * G3 + HID_D + j];
  const float nxb = bx[k * G3 + 2 * HID_D + j];
  const float nhb = bh[k * G3 + 2 * HID_D + j];

  const float* xrow = x + (size_t)brow * T_SZ * IN_D;

  for (int t = 0; t < T_SZ; ++t) {
    const float* hcur = (t & 1) ? hb1 : hb0;
    float*       hnxt = (t & 1) ? hb0 : hb1;

    v8f accR = zero8(), accZ = zero8(), accNX = zero8(), accNH = zero8();

    // Preload ALL gh A-operands and the h_old scalars first: their L2
    // latency (producers were other WGPs last step) hides under gx WMMAs.
    const float* ah = hcur + ((size_t)k * B_SZ + brow) * HID_D;
    v16bf ahv[8];
#pragma unroll
    for (int ks = 0; ks < 8; ++ks) ahv[ks] = load_a_row(ah + ks * 32, half);

    const float* hco = hcur + (size_t)k * B_SZ * HID_D;
    float ho[8];
#pragma unroll
    for (int v = 0; v < 8; ++v) {
      const int row = rowHalf * 64 + wave * 16 + v + 8 * half;
      ho[v] = hco[(size_t)row * HID_D + j];
    }

    const float* ax = xrow + (size_t)t * IN_D;
    if (t + 1 < T_SZ) {                       // pull next x row out of HBM
#pragma unroll
      for (int pf = 0; pf < 8; ++pf)
        __builtin_prefetch(ax + IN_D + 32 * pf, 0, 0);
    }

    // gx = x_t @ Wx[k]  (K = IN = 256 -> 8 WMMA k-steps of 32)
#pragma unroll
    for (int ks = 0; ks < 8; ++ks) {
      const v16bf a = load_a_row(ax + ks * 32, half);
      accR  = wmma_bf16(a, breg[0][0][ks], accR);
      accZ  = wmma_bf16(a, breg[0][1][ks], accZ);
      accNX = wmma_bf16(a, breg[0][2][ks], accNX);
    }

    // gh = h[k] @ Wh[k]  (r,z fuse into same accumulators; nh kept apart).
    // Pure register-resident WMMA chain.
#pragma unroll
    for (int ks = 0; ks < 8; ++ks) {
      accR  = wmma_bf16(ahv[ks], breg[1][0][ks], accR);
      accZ  = wmma_bf16(ahv[ks], breg[1][1][ks], accZ);
      accNH = wmma_bf16(ahv[ks], breg[1][2][ks], accNH);
    }

    // Gates + state update. D layout: element v -> row M = v + 8*half.
    float* hno = hnxt + (size_t)k * B_SZ * HID_D;
#pragma unroll
    for (int v = 0; v < 8; ++v) {
      const int row = rowHalf * 64 + wave * 16 + v + 8 * half;
      const float r  = sigmoidf_(accR[v] + rb);
      const float z  = sigmoidf_(accZ[v] + zb);
      const float n  = tanhf(accNX[v] + nxb + r * (accNH[v] + nhb));
      hno[(size_t)row * HID_D + j] = (1.0f - z) * n + z * ho[v];
    }

    grid_barrier(bar, bar + 1, (unsigned)(t + 1));
  }
}

// ---------------------------------------------------------------------------
// out = elu( (h0+h1+h2) @ W_fc + b_fc )   [128,256]x[256,256] via WMMA.
// Grid: 16 WGs of 256 threads; WG n owns 16 output columns, wave m owns rows
// 16m..16m+15.
// ---------------------------------------------------------------------------
__global__ __launch_bounds__(256)
void cru_fc(const float* __restrict__ hf,    // final h [3,B,H]
            const float* __restrict__ Wfc,   // [HID,OUT]
            const float* __restrict__ bfc,   // [OUT]
            float* __restrict__ out) {       // [B,OUT]
  const int tid  = threadIdx.x;
  const int lane = tid & 31;
  const int wave = tid >> 5;
  const int half = lane >> 4;
  const int nlo  = lane & 15;
  const int c0   = blockIdx.x * 16;
  const int arow = wave * 16 + nlo;
  const int col  = c0 + nlo;

  v8f acc = zero8();
#pragma unroll
  for (int ks = 0; ks < 8; ++ks) {
    v16bf a;
#pragma unroll
    for (int v = 0; v < 8; ++v) {
      const int kk = ks * 32 + ((v < 4) ? 2 * v : 16 + 2 * (v - 4)) + 8 * half;
      const float2 s0 = *(const float2*)(hf + ((size_t)0 * B_SZ + arow) * HID_D + kk);
      const float2 s1 = *(const float2*)(hf + ((size_t)1 * B_SZ + arow) * HID_D + kk);
      const float2 s2 = *(const float2*)(hf + ((size_t)2 * B_SZ + arow) * HID_D + kk);
      a[2 * v]     = (__bf16)(s0.x + s1.x + s2.x);   // pooled = sum over k
      a[2 * v + 1] = (__bf16)(s0.y + s1.y + s2.y);
    }
    v16bf b;
#pragma unroll
    for (int e = 0; e < 16; ++e) {
      const int K = ks * 32 + half * 8 + e + ((e >= 8) ? 8 : 0);
      b[e] = (__bf16)Wfc[(size_t)K * OUT_D + col];
    }
    acc = wmma_bf16(a, b, acc);
  }

  const float bias = bfc[col];
#pragma unroll
  for (int v = 0; v < 8; ++v) {
    const int row = wave * 16 + v + 8 * half;
    const float val = acc[v] + bias;
    out[(size_t)row * OUT_D + col] = (val > 0.0f) ? val : (__expf(val) - 1.0f);
  }
}

// feature[k, j] = mean over batch of h_final[k, :, j]
__global__ void cru_feature(const float* __restrict__ hf, float* __restrict__ feat) {
  const int k = blockIdx.x;
  const int j = threadIdx.x;
  float s = 0.0f;
  for (int b = 0; b < B_SZ; ++b)
    s += hf[((size_t)k * B_SZ + b) * HID_D + j];
  feat[k * HID_D + j] = s * (1.0f / (float)B_SZ);
}

// ---------------------------------------------------------------------------
extern "C" void kernel_launch(void* const* d_in, const int* in_sizes, int n_in,
                              void* d_out, int out_size, void* d_ws, size_t ws_size,
                              hipStream_t stream) {
  const float* x   = (const float*)d_in[0];   // [B,T,IN]
  const float* h0  = (const float*)d_in[1];   // [1,3,B,H]
  const float* Wx  = (const float*)d_in[2];   // [3,IN,768]
  const float* Wh  = (const float*)d_in[3];   // [3,HID,768]
  const float* bx  = (const float*)d_in[4];   // [3,768]
  const float* bh  = (const float*)d_in[5];   // [3,768]
  const float* Wfc = (const float*)d_in[6];   // [HID,OUT]
  const float* bfc = (const float*)d_in[7];   // [OUT]
  float* out = (float*)d_out;                 // [B,OUT] then feature [3,H]

  float* hb0 = (float*)d_ws;                  // ws: 2 h buffers + barrier
  float* hb1 = hb0 + HELEMS;
  unsigned* bar = (unsigned*)(hb1 + HELEMS);

  cru_init<<<(HELEMS + 255) / 256, 256, 0, stream>>>(h0, hb0, bar);
  cru_persistent<<<NWG, 128, 0, stream>>>(x, Wx, Wh, bx, bh, hb0, hb1, bar);
  // T_SZ is even: final state lands in hb0 (last step t=2047 writes hb0).
  cru_fc<<<OUT_D / 16, 256, 0, stream>>>(hb0, Wfc, bfc, out);
  cru_feature<<<3, HID_D, 0, stream>>>(hb0, out + (size_t)B_SZ * OUT_D);
  (void)in_sizes; (void)n_in; (void)out_size; (void)ws_size;
}
